// Summarizer_20083267076575
// MI455X (gfx1250) — matrix operand order repaired and checked by reading, not compiled, not run on hardware
//
#include <hip/hip_runtime.h>
#include <hip/hip_bf16.h>
#include <cstddef>
#include <cstdint>

// Sizes: B=16 S=32 W=32 T=64 H2=512 Hf=256 F=128 A=128 V=32000, SW=1024
typedef __attribute__((ext_vector_type(16))) _Float16 v16h;
typedef __attribute__((ext_vector_type(8)))  float    v8f;

#define DEV __device__ __forceinline__

// ---------------- WMMA helpers (gfx1250, wave32) ------------------------------
DEV v8f wmma32(v16h a, v16h b, v8f c) {
  return __builtin_amdgcn_wmma_f32_16x16x32_f16(false, a, false, b, (short)0, c, false, false);
}

// A fragment (16x32 f16) from f32 row-major, pre-offset to (row0,k0); lda in floats.
DEV v16h frag_a(const float* A, int lda, int lane) {
  int r = lane & 15, koff = (lane >> 4) * 8;
  v16h f;
#pragma unroll
  for (int e = 0; e < 16; ++e) {
    int k = koff + (e < 8 ? e : e + 8);
    f[e] = (_Float16)A[(size_t)r * lda + k];
  }
  return f;
}

// A fragment from f16 row-major (two contiguous 16B runs per lane -> 2x b128).
DEV v16h frag_a_h(const _Float16* A, int lda, int lane) {
  int r = lane & 15, koff = (lane >> 4) * 8;
  v16h f;
#pragma unroll
  for (int e = 0; e < 16; ++e) {
    int k = koff + (e < 8 ? e : e + 8);
    f[e] = A[(size_t)r * lda + k];
  }
  return f;
}

// A fragment for logical column-concat [P0 | P1] split at `split` (split % 32 == 0).
DEV v16h frag_a_cat2(const float* P0, int lda0, const float* P1, int lda1,
                     int split, int k0, int lane) {
  int r = lane & 15, koff = (lane >> 4) * 8;
  v16h f;
#pragma unroll
  for (int e = 0; e < 16; ++e) {
    int k = k0 + koff + (e < 8 ? e : e + 8);
    float v = (k < split) ? P0[(size_t)r * lda0 + k] : P1[(size_t)r * lda1 + (k - split)];
    f[e] = (_Float16)v;
  }
  return f;
}

// A fragment for [emb[summary[b,t]] | ctx] (split at 128).
DEV v16h frag_a_embctx(const float* emb, const int* summary, int t, const float* ctx,
                       int k0, int lane) {
  int b = lane & 15, koff = (lane >> 4) * 8;
  int tok = summary[b * 64 + t];
  v16h f;
#pragma unroll
  for (int e = 0; e < 16; ++e) {
    int k = k0 + koff + (e < 8 ? e : e + 8);
    float v = (k < 128) ? emb[(size_t)tok * 128 + k] : ctx[(size_t)b * 512 + (k - 128)];
    f[e] = (_Float16)v;
  }
  return f;
}

// Pre-swizzled B fragment: frag = nt*nKb + kb; 32 lanes x 16 halves contiguous.
DEV v16h frag_b_pre(const _Float16* base, int frag, int lane) {
  return *(const v16h*)(base + ((size_t)frag * 32 + lane) * 16);
}

DEV float sigm(float x) { return 1.f / (1.f + expf(-x)); }

DEV float wave_sum(float v) {
#pragma unroll
  for (int off = 16; off > 0; off >>= 1) v += __shfl_xor(v, off, 32);
  return v;
}

// ---------------- text_states f32 -> f16 copy (once) --------------------------
__global__ void k_cvt_ts(const float* __restrict__ src, _Float16* __restrict__ dst) {
  size_t i = ((size_t)blockIdx.x * 256 + threadIdx.x) * 4;
  float4 v = *(const float4*)(src + i);
  dst[i + 0] = (_Float16)v.x; dst[i + 1] = (_Float16)v.y;
  dst[i + 2] = (_Float16)v.z; dst[i + 3] = (_Float16)v.w;
}

// ---------------- Weight swizzle: row-major f32 (K x ldb) -> B-fragment f16 ---
__global__ void k_swz_b(const float* __restrict__ P0, const float* __restrict__ P1,
                        int split, int ldb, int ncols, int K, int ntiles,
                        _Float16* __restrict__ dst) {
  int lane = threadIdx.x & 31, wave = threadIdx.x >> 5;
  int nKb = K >> 5;
  int frag = blockIdx.x * 8 + wave;
  if (frag >= ntiles * nKb) return;
  int nt = frag / nKb, kb = frag - nt * nKb;
  int n = nt * 16 + (lane & 15); if (n >= ncols) n = ncols - 1;
  int kbase = kb * 32 + ((lane >> 4) << 3);
  _Float16* out = dst + ((size_t)frag * 32 + lane) * 16;
#pragma unroll
  for (int e = 0; e < 16; ++e) {
    int k = kbase + (e < 8 ? e : e + 8);
    float v = (k < split) ? P0[(size_t)k * ldb + n] : P1[(size_t)(k - split) * ldb + n];
    out[e] = (_Float16)v;
  }
}

// ---------------- Precompute: ws_a = ws @ Wa_word + b_a (16384x512 * 512x128) -
__global__ void k_wsa(const _Float16* __restrict__ tsh, const _Float16* __restrict__ fB,
                      const float* __restrict__ ba, float* __restrict__ wsa) {
  int lane = threadIdx.x & 31, wave = threadIdx.x >> 5;
  int mt = blockIdx.x;            // 0..1023
  const _Float16* A0 = tsh + (size_t)mt * 16 * 512;
  v8f acc = {};
  for (int kb = 0; kb < 16; ++kb)
    acc = wmma32(frag_a_h(A0 + kb * 32, 512, lane), frag_b_pre(fB, wave * 16 + kb, lane), acc);
  int col = wave * 16 + (lane & 15);
  int r0 = mt * 16 + ((lane >> 4) * 8);
  float bb = ba[col];
#pragma unroll
  for (int e = 0; e < 8; ++e)
    wsa[(size_t)(r0 + e) * 128 + col] = acc[e] + bb;
}

// ---------------- Init carried state ------------------------------------------
__global__ void k_init(const float* state, float* h, float* c, float* ctxc,
                       float* cov, float* loss) {
  int idx = blockIdx.x * 256 + threadIdx.x;
  if (idx < 16384) cov[idx] = 0.f;
  if (idx < 8192) {
    int b = idx >> 9, j = idx & 511;
    h[idx]    = state[(size_t)b * 1024 + j];
    c[idx]    = state[(size_t)b * 1024 + 512 + j];
    ctxc[idx] = 0.f;
  }
  if (idx < 16) loss[idx] = 0.f;
}

// ---------------- x = [emb|ctx] @ W_cc + b_cc  (16x640 * 640x128) -------------
__global__ void k_xgemm(const float* emb, const int* summary, const float* ctxc,
                        const _Float16* fB, const float* bcc, float* x, int t) {
  int lane = threadIdx.x & 31, wave = threadIdx.x >> 5;
  v8f acc = {};
  for (int kb = 0; kb < 20; ++kb)
    acc = wmma32(frag_a_embctx(emb, summary, t, ctxc, kb * 32, lane),
                 frag_b_pre(fB, wave * 20 + kb, lane), acc);
  int col = wave * 16 + (lane & 15), r0 = (lane >> 4) * 8;
  float bb = bcc[col];
#pragma unroll
  for (int e = 0; e < 8; ++e) x[(size_t)(r0 + e) * 128 + col] = acc[e] + bb;
}

// ---------------- gates = [x|h] @ [Wx;Wh] + b  (16x640 * 640x2048) ------------
__global__ void k_gates(const float* x, const float* h, const _Float16* fB,
                        const float* bl, float* gates) {
  int lane = threadIdx.x & 31, wave = threadIdx.x >> 5;
  int wid = blockIdx.x * 8 + wave;     // 0..63
  int nt0 = wid * 2, nt1 = nt0 + 1;    // 0..127
  v8f a0 = {}, a1 = {};
  for (int kb = 0; kb < 20; ++kb) {
    __builtin_prefetch(fB + ((size_t)(nt0 * 20 + kb + 1) * 32) * 16, 0, 1);
    v16h af = frag_a_cat2(x, 128, h, 512, 128, kb * 32, lane);
    a0 = wmma32(af, frag_b_pre(fB, nt0 * 20 + kb, lane), a0);
    a1 = wmma32(af, frag_b_pre(fB, nt1 * 20 + kb, lane), a1);
  }
  int nl = lane & 15, r0 = (lane >> 4) * 8;
  int c0 = nt0 * 16 + nl, c1 = nt1 * 16 + nl;
  float b0 = bl[c0], b1 = bl[c1];
#pragma unroll
  for (int e = 0; e < 8; ++e) {
    gates[(size_t)(r0 + e) * 2048 + c0] = a0[e] + b0;
    gates[(size_t)(r0 + e) * 2048 + c1] = a1[e] + b1;
  }
}

// ---------------- LSTM pointwise; carry h,c scaled by valid -------------------
__global__ void k_lstm(const float* gates, float* h, float* c, float* outb,
                       const int* slen, int t) {
  int idx = blockIdx.x * 256 + threadIdx.x;   // 0..8191
  int b = idx >> 9, j = idx & 511;
  const float* g = gates + (size_t)b * 2048;
  float gi = g[j], gf = g[512 + j], gg = g[1024 + j], go = g[1536 + j];
  float cn = sigm(gf) * c[idx] + sigm(gi) * tanhf(gg);
  float o  = sigm(go) * tanhf(cn);
  float valid = (slen[b] - t - 1 > 0) ? 1.f : 0.f;
  outb[idx] = o;
  h[idx] = o * valid;
  c[idx] = cn * valid;
}

// ---------------- q = out @ Wa_out  (16x512 * 512x128) ------------------------
__global__ void k_qgemm(const float* outb, const _Float16* fB, float* q) {
  int lane = threadIdx.x & 31, wave = threadIdx.x >> 5;
  v8f acc = {};
  for (int kb = 0; kb < 16; ++kb)
    acc = wmma32(frag_a(outb + kb * 32, 512, lane), frag_b_pre(fB, wave * 16 + kb, lane), acc);
  int col = wave * 16 + (lane & 15), r0 = (lane >> 4) * 8;
#pragma unroll
  for (int e = 0; e < 8; ++e) q[(size_t)(r0 + e) * 128 + col] = acc[e];
}

// ---------------- scores[b,n] = tanh(ws_a + q + cov*wa_cov) . v_a -------------
__global__ void k_scores(const float* wsa, const float* q, const float* cov,
                         const float* Wa, const float* va, float* scores) {
  int lane = threadIdx.x & 31, wave = threadIdx.x >> 5;
  int wid = blockIdx.x * 8 + wave;            // (b,n)
  int b = wid >> 10, n = wid & 1023;
  const float* wacov = Wa + (size_t)1024 * 128;
  float cv = cov[(size_t)b * 1024 + n];
  const float* wr = wsa + ((size_t)b * 1024 + n) * 128;
  const float* qr = q + (size_t)b * 128;
  float acc = 0.f;
#pragma unroll
  for (int i = 0; i < 4; ++i) {
    int a = lane + i * 32;
    acc += tanhf(wr[a] + qr[a] + cv * wacov[a]) * va[a];
  }
  acc = wave_sum(acc);
  if (lane == 0) scores[(size_t)b * 1024 + n] = acc;
}

// ---------------- masked softmax + coverage update + covloss ------------------
__global__ void k_softmax(const float* scores, const int* tlen, const int* slen, int t,
                          float* attn, float* cov, float* cvl) {
  __shared__ float red[256];
  int b = blockIdx.x, tid = threadIdx.x;
  float valid = (slen[b] - t - 1 > 0) ? 1.f : 0.f;
  float s[4];
  float m = -3.4e38f;
#pragma unroll
  for (int i = 0; i < 4; ++i) {
    int n = tid + i * 256;
    int sent = n >> 5, w = n & 31;
    float v = (w < tlen[b * 32 + sent]) ? scores[(size_t)b * 1024 + n] : -1.0e9f;
    s[i] = v;
    m = fmaxf(m, v);
  }
  red[tid] = m; __syncthreads();
  for (int off = 128; off > 0; off >>= 1) { if (tid < off) red[tid] = fmaxf(red[tid], red[tid + off]); __syncthreads(); }
  m = red[0]; __syncthreads();
  float sum = 0.f;
#pragma unroll
  for (int i = 0; i < 4; ++i) { s[i] = expf(s[i] - m); sum += s[i]; }
  red[tid] = sum; __syncthreads();
  for (int off = 128; off > 0; off >>= 1) { if (tid < off) red[tid] += red[tid + off]; __syncthreads(); }
  float inv = 1.f / red[0];
  __syncthreads();
  float cl = 0.f;
#pragma unroll
  for (int i = 0; i < 4; ++i) {
    int n = tid + i * 256;
    float a = s[i] * inv;
    attn[(size_t)b * 1024 + n] = a;
    float cv = cov[(size_t)b * 1024 + n];
    cl += fminf(cv, a);                        // covloss on pre-update coverage
    cov[(size_t)b * 1024 + n] = cv + a * valid;
  }
  red[tid] = cl; __syncthreads();
  for (int off = 128; off > 0; off >>= 1) { if (tid < off) red[tid] += red[tid + off]; __syncthreads(); }
  if (tid == 0) cvl[b] = red[0];
}

// ---------------- ctx4 = attn @ cat_states (f16 ts; 2 cols/thread) ------------
__global__ void k_ctx4(const float* attn, const _Float16* tsh, const float* ss,
                       float* ctx4) {
  __shared__ float sat[1024];
  __shared__ float ssum[32];
  int blk = blockIdx.x;               // 0..63
  bool second = blk >= 32; if (second) blk -= 32;
  int b = blk >> 1;
  int d0 = (blk & 1) * 256 + threadIdx.x * 2;   // 0..510
  for (int i = threadIdx.x; i < 1024; i += 128) sat[i] = attn[(size_t)b * 1024 + i];
  __syncthreads();
  if (!second) {
    const _Float16* A = tsh + (size_t)b * 1024 * 512 + d0;
    float a0 = 0.f, a1 = 0.f;
    for (int n = 0; n < 1024; ++n) {
      float w = sat[n];
      a0 += w * (float)A[(size_t)n * 512];
      a1 += w * (float)A[(size_t)n * 512 + 1];
    }
    ctx4[(size_t)b * 1024 + d0]     = a0;
    ctx4[(size_t)b * 1024 + d0 + 1] = a1;
  } else {
    if (threadIdx.x < 32) {
      float sv = 0.f;
      for (int w = 0; w < 32; ++w) sv += sat[threadIdx.x * 32 + w];
      ssum[threadIdx.x] = sv;
    }
    __syncthreads();
    float a0 = 0.f, a1 = 0.f;
    for (int sI = 0; sI < 32; ++sI) {
      float w = ssum[sI];
      const float* S = ss + ((size_t)b * 32 + sI) * 512 + d0;
      a0 += w * S[0];
      a1 += w * S[1];
    }
    ctx4[(size_t)b * 1024 + 512 + d0]     = a0;
    ctx4[(size_t)b * 1024 + 512 + d0 + 1] = a1;
  }
}

// ---------------- ctx_new = ctx4 @ W_rc + b_rc; carry = ctx_new*valid ---------
__global__ void k_rc(const float* ctx4, const _Float16* fB, const float* brc,
                     const int* slen, int t, float* ctxn, float* ctxc) {
  int lane = threadIdx.x & 31, wave = threadIdx.x >> 5;
  int nt = blockIdx.x * 8 + wave;     // 0..31
  v8f acc = {};
  for (int kb = 0; kb < 32; ++kb)
    acc = wmma32(frag_a(ctx4 + kb * 32, 1024, lane), frag_b_pre(fB, nt * 32 + kb, lane), acc);
  int col = nt * 16 + (lane & 15), r0 = (lane >> 4) * 8;
  float bb = brc[col];
#pragma unroll
  for (int e = 0; e < 8; ++e) {
    int r = r0 + e;
    float v = acc[e] + bb;
    float valid = (slen[r] - t - 1 > 0) ? 1.f : 0.f;
    ctxn[(size_t)r * 512 + col] = v;
    ctxc[(size_t)r * 512 + col] = v * valid;
  }
}

// ---------------- hid = relu([ctx_new|out] @ W1 + b1)  (16x1024 * 1024x256) ---
__global__ void k_hid(const float* ctxn, const float* outb, const _Float16* fB,
                      const float* b1, float* hid) {
  int lane = threadIdx.x & 31, wave = threadIdx.x >> 5;
  int nt = blockIdx.x * 8 + wave;     // 0..15
  v8f acc = {};
  for (int kb = 0; kb < 32; ++kb)
    acc = wmma32(frag_a_cat2(ctxn, 512, outb, 512, 512, kb * 32, lane),
                 frag_b_pre(fB, nt * 32 + kb, lane), acc);
  int col = nt * 16 + (lane & 15), r0 = (lane >> 4) * 8;
  float bb = b1[col];
#pragma unroll
  for (int e = 0; e < 8; ++e)
    hid[(size_t)(r0 + e) * 256 + col] = fmaxf(acc[e] + bb, 0.f);
}

// ---------------- logits = hid @ W2 + b2  (16x256 * 256x32001) ----------------
// 2 N-tiles per wave (A-fragment reuse, 2 B-streams in flight).
__global__ void k_logits(const float* hid, const _Float16* fB, const float* b2,
                         float* logits) {
  int lane = threadIdx.x & 31, wave = threadIdx.x >> 5;
  int wid = blockIdx.x * 8 + wave;      // 0..1007
  int nt0 = wid * 2, nt1 = nt0 + 1;
  if (nt0 >= 2001) return;              // wave-uniform: EXEC full at WMMA
  bool has1 = (nt1 < 2001);
  v8f a0 = {}, a1 = {};
#pragma unroll
  for (int kb = 0; kb < 8; ++kb) {
    int pf = nt0 * 8 + kb + 16; if (pf > 16007) pf = 16007;   // clamped prefetch
    __builtin_prefetch(fB + (size_t)pf * 512, 0, 1);
    v16h af = frag_a(hid + kb * 32, 256, lane);
    a0 = wmma32(af, frag_b_pre(fB, nt0 * 8 + kb, lane), a0);
    if (has1) a1 = wmma32(af, frag_b_pre(fB, nt1 * 8 + kb, lane), a1);
  }
  int nl = lane & 15, r0 = (lane >> 4) * 8;
  int c0 = nt0 * 16 + nl;
  if (c0 < 32001) {
    float bb = b2[c0];
#pragma unroll
    for (int e = 0; e < 8; ++e)
      logits[(size_t)(r0 + e) * 32001 + c0] = a0[e] + bb;
  }
  if (has1) {
    int c1 = nt1 * 16 + nl;
    float bb = b2[c1];
#pragma unroll
    for (int e = 0; e < 8; ++e)
      logits[(size_t)(r0 + e) * 32001 + c1] = a1[e] + bb;
  }
}

// ---------------- logsumexp + NLL + coverage loss accumulate ------------------
__global__ void k_loss(const float* logits, const int* summary, const int* slen,
                       const float* cvl, int t, float* loss) {
  __shared__ float red[256];
  int b = blockIdx.x, tid = threadIdx.x;
  const float* L = logits + (size_t)b * 32001;
  float m = -3.4e38f;
  for (int i = tid; i < 32001; i += 256) m = fmaxf(m, L[i]);
  red[tid] = m; __syncthreads();
  for (int off = 128; off > 0; off >>= 1) { if (tid < off) red[tid] = fmaxf(red[tid], red[tid + off]); __syncthreads(); }
  m = red[0]; __syncthreads();
  float s = 0.f;
  for (int i = tid; i < 32001; i += 256) s += expf(L[i] - m);
  red[tid] = s; __syncthreads();
  for (int off = 128; off > 0; off >>= 1) { if (tid < off) red[tid] += red[tid + off]; __syncthreads(); }
  if (tid == 0) {
    int tok = summary[b * 64 + t + 1];
    int tgt = (tok == -1) ? 32000 : tok;
    float logp = L[tgt] - m - logf(red[0]);
    float valid = (slen[b] - t - 1 > 0) ? 1.f : 0.f;
    loss[b] += valid * (-logp + cvl[b]);
  }
}

__global__ void k_final(const float* loss, const int* slen, float* out) {
  int b = threadIdx.x;
  if (b < 16) out[b] = loss[b] / ((float)slen[b] - 1.f);
}

// =============================================================================
extern "C" void kernel_launch(void* const* d_in, const int* in_sizes, int n_in,
                              void* d_out, int out_size, void* d_ws, size_t ws_size,
                              hipStream_t stream) {
  (void)in_sizes; (void)n_in; (void)out_size; (void)ws_size;
  const float* ts   = (const float*)d_in[0];   // text_states (16,32,32,512)
  const float* ss   = (const float*)d_in[1];   // sentence_states (16,32,512)
  const int*   tlen = (const int*)  d_in[2];   // text_length (16,32)
  const float* st   = (const float*)d_in[3];   // state (16,2,512)
  const int*   summ = (const int*)  d_in[4];   // summary (16,64)
  const int*   slen = (const int*)  d_in[5];   // summary_length (16,)
  const float* emb  = (const float*)d_in[6];   // (32001,128)
  const float* Wcc  = (const float*)d_in[7];   // (640,128)
  const float* bcc  = (const float*)d_in[8];
  const float* Wx   = (const float*)d_in[9];   // (128,2048)
  const float* Wh   = (const float*)d_in[10];  // (512,2048)
  const float* bl   = (const float*)d_in[11];
  const float* Wa   = (const float*)d_in[12];  // (1025,128)
  const float* ba   = (const float*)d_in[13];
  const float* va   = (const float*)d_in[14];
  const float* Wrc  = (const float*)d_in[15];  // (1024,512)
  const float* brc  = (const float*)d_in[16];
  const float* W1   = (const float*)d_in[17];  // (1024,256)
  const float* b1   = (const float*)d_in[18];
  const float* W2   = (const float*)d_in[19];  // (256,32001)
  const float* b2   = (const float*)d_in[20];

  // ---- f32 scratch region (~11 MB) ----
  float* ws     = (float*)d_ws;
  float* wsa    = ws;                         // 2,097,152
  float* logits = wsa + 2097152;              // 512,016
  float* h      = logits + 512016;            // 8192 each below
  float* c      = h + 8192;
  float* outb   = c + 8192;
  float* ctxn   = outb + 8192;
  float* ctxc   = ctxn + 8192;
  float* xbuf   = ctxc + 8192;                // 2048
  float* q      = xbuf + 2048;
  float* gates  = q + 2048;                   // 32768
  float* scores = gates + 32768;              // 16384 each below
  float* attn   = scores + 16384;
  float* cov    = attn + 16384;
  float* ctx4   = cov + 16384;
  float* hid    = ctx4 + 16384;               // 4096
  float* cvl    = hid + 4096;                 // 16
  float* loss   = cvl + 16;                   // 16

  // ---- f16 region (~38 MB): pre-swizzled weight fragments + f16 text_states --
  uintptr_t pa = ((uintptr_t)(loss + 16) + 63) & ~(uintptr_t)63;
  _Float16* fWaW = (_Float16*)pa;             //  8*16*512 =    65,536
  _Float16* fWcc = fWaW + 65536;              //  8*20*512 =    81,920
  _Float16* fWxh = fWcc + 81920;              // 128*20*512 = 1,310,720
  _Float16* fWaO = fWxh + 1310720;            //  8*16*512 =    65,536
  _Float16* fWrc = fWaO + 65536;              // 32*32*512 =   524,288
  _Float16* fW1  = fWrc + 524288;             // 16*32*512 =   262,144
  _Float16* fW2  = fW1  + 262144;             // 2001*8*512 = 8,196,096
  _Float16* tsh  = fW2  + 8196096;            // 8,388,608 (16.8 MB)

  // One-time conversions (deterministic each launch).
  k_cvt_ts<<<8192, 256, 0, stream>>>(ts, tsh);
  //            P0            P1            split ldb    ncols  K    ntiles dst
  k_swz_b<<<16,   256, 0, stream>>>(Wa,           Wa,           512,  128,   128,   512,  8,    fWaW);
  k_swz_b<<<20,   256, 0, stream>>>(Wcc,          Wcc,          640,  128,   128,   640,  8,    fWcc);
  k_swz_b<<<320,  256, 0, stream>>>(Wx,           Wh,           128,  2048,  2048,  640,  128,  fWxh);
  k_swz_b<<<16,   256, 0, stream>>>(Wa + 65536,   Wa + 65536,   512,  128,   128,   512,  8,    fWaO);
  k_swz_b<<<128,  256, 0, stream>>>(Wrc,          Wrc,          1024, 512,   512,   1024, 32,   fWrc);
  k_swz_b<<<64,   256, 0, stream>>>(W1,           W1,           1024, 256,   256,   1024, 16,   fW1);
  k_swz_b<<<2001, 256, 0, stream>>>(W2,           W2,           256,  32001, 32001, 256,  2001, fW2);

  k_wsa <<<1024, 256, 0, stream>>>(tsh, fWaW, ba, wsa);
  k_init<<<64,   256, 0, stream>>>(st, h, c, ctxc, cov, loss);

  for (int t = 0; t < 63; ++t) {
    k_xgemm  <<<1,    256, 0, stream>>>(emb, summ, ctxc, fWcc, bcc, xbuf, t);
    k_gates  <<<8,    256, 0, stream>>>(xbuf, h, fWxh, bl, gates);
    k_lstm   <<<32,   256, 0, stream>>>(gates, h, c, outb, slen, t);
    k_qgemm  <<<1,    256, 0, stream>>>(outb, fWaO, q);
    k_scores <<<2048, 256, 0, stream>>>(wsa, q, cov, Wa, va, scores);
    k_softmax<<<16,   256, 0, stream>>>(scores, tlen, slen, t, attn, cov, cvl);
    k_ctx4   <<<64,   128, 0, stream>>>(attn, tsh, ss, ctx4);
    k_rc     <<<4,    256, 0, stream>>>(ctx4, fWrc, brc, slen, t, ctxn, ctxc);
    k_hid    <<<2,    256, 0, stream>>>(ctxn, outb, fW1, b1, hid);
    k_logits <<<126,  256, 0, stream>>>(hid, fW2, b2, logits);
    k_loss   <<<16,   256, 0, stream>>>(logits, summ, slen, cvl, t, loss);
  }
  k_final<<<1, 16, 0, stream>>>(loss, slen, (float*)d_out);
}
